// Spatio_Tmp_Embed_75977971466791
// MI455X (gfx1250) — matrix operand order repaired and checked by reading, not compiled, not run on hardware
//
#include <hip/hip_runtime.h>
#include <stdint.h>

#define NN 50000
#define NE 800000

typedef __attribute__((ext_vector_type(16))) _Float16 v16h;
typedef __attribute__((ext_vector_type(8)))  float    v8f;

// ---------------- WMMA helpers (gfx1250, wave32) ----------------
__device__ __forceinline__ v8f wmma16(v16h a, v16h b, v8f c) {
  return __builtin_amdgcn_wmma_f32_16x16x32_f16(false, a, false, b, (short)0, c,
                                                false, false);
}

// A fragment (16x32 f16, row-major source): lane holds row (lane&15),
// K pairs: V0..3 -> K{0..7}+8*half, V4..7 -> K{16..23}+8*half
__device__ __forceinline__ v16h frag_a(const _Float16* row, int hf) {
  union { v16h h; unsigned u[8]; } r;
  const unsigned* p = (const unsigned*)row;
  const int h4 = hf * 4;
#pragma unroll
  for (int v = 0; v < 8; ++v) r.u[v] = p[(v < 4 ? 0 : 8) + h4 + (v & 3)];
  return r.h;
}
// B fragment from B^T (row-major [N,K]): lane holds col (lane&15),
// K pairs: V0..3 -> K{0..7}+8*half, V4..7 -> K{16..23}+8*half
__device__ __forceinline__ v16h frag_b(const _Float16* row, int hf) {
  union { v16h h; unsigned u[8]; } r;
  const unsigned* p = (const unsigned*)row;
#pragma unroll
  for (int v = 0; v < 8; ++v) r.u[v] = p[(v < 4 ? 0 : 8) + hf * 4 + (v & 3)];
  return r.h;
}

// fast pow for clamped-positive base: exp(p*log(v)); exact fast path for p==1
__device__ __forceinline__ float fast_pow_pos(float v, float p) {
  if (p == 1.0f) return v;             // wave-uniform branch (p is a scalar param)
  return __expf(p * __logf(v));
}

// ---------------- weight convert: f32 [K,Nc] -> f16 [Nc,K] ----------------
__global__ void wconv_kernel(const float* __restrict__ W, _Float16* __restrict__ WT,
                             int K, int Nc) {
  int i = blockIdx.x * blockDim.x + threadIdx.x;
  if (i < K * Nc) {
    int k = i / Nc, n = i - k * Nc;
    WT[(size_t)n * K + k] = (_Float16)W[i];
  }
}

// ---------------- edge features: ef = LN(concat(ta@Wt+bt, aux@Wa+ba)) ------
__global__ void ef_kernel(const float* __restrict__ ta, const float* __restrict__ aux,
                          const float* __restrict__ Wt, const float* __restrict__ bt,
                          const float* __restrict__ Wa, const float* __restrict__ ba,
                          const float* __restrict__ ge, const float* __restrict__ be,
                          _Float16* __restrict__ efh) {
  int g = threadIdx.x >> 6, j = threadIdx.x & 63;
  int e = blockIdx.x * 4 + g;
  __shared__ float sta[4][64];
  __shared__ float red[4][64];
  sta[g][j] = ta[(size_t)e * 64 + j];
  __syncthreads();
  float val;
  if (j < 32) {
    float s = bt[j];
#pragma unroll 8
    for (int k = 0; k < 64; ++k) s += sta[g][k] * Wt[k * 32 + j];
    val = s;
  } else {
    int c = j - 32;
    val = aux[(size_t)e * 2] * Wa[c] + aux[(size_t)e * 2 + 1] * Wa[32 + c] + ba[c];
  }
  red[g][j] = val; __syncthreads();
  for (int s = 32; s > 0; s >>= 1) { if (j < s) red[g][j] += red[g][j + s]; __syncthreads(); }
  float mean = red[g][0] * (1.f / 64.f);
  __syncthreads();
  float dv = val - mean;
  red[g][j] = dv * dv; __syncthreads();
  for (int s = 32; s > 0; s >>= 1) { if (j < s) red[g][j] += red[g][j + s]; __syncthreads(); }
  float var = red[g][0] * (1.f / 64.f);
  float o = dv * rsqrtf(var + 1e-5f) * ge[j] + be[j];
  efh[(size_t)e * 64 + j] = (_Float16)o;
}

// ------- fused msg GEMM + power-mean scatter: 16 edges x 128 cols/block ----
__global__ void msg_agg_kernel(const _Float16* __restrict__ efh,
                               const _Float16* __restrict__ WeT,
                               const float* __restrict__ be,
                               const float* __restrict__ x,
                               const int* __restrict__ src, const int* __restrict__ dst,
                               const float* __restrict__ p_pow,
                               float* __restrict__ agg, float* __restrict__ cnt) {
  int wv = threadIdx.x >> 5;
  int lane = threadIdx.x & 31, hf = lane >> 4, n15 = lane & 15;
  int ebase = blockIdx.x * 16;
  const _Float16* arow = efh + (size_t)(ebase + n15) * 64;
  const _Float16* brow = WeT + (size_t)(wv * 16 + n15) * 64;
  v8f acc = {0.f, 0.f, 0.f, 0.f, 0.f, 0.f, 0.f, 0.f};
#pragma unroll
  for (int k = 0; k < 64; k += 32)
    acc = wmma16(frag_a(arow + k, hf), frag_b(brow + k, hf), acc);
  float p = p_pow[0];
#pragma unroll
  for (int r = 0; r < 8; ++r) {
    int e = ebase + r + hf * 8;
    int n = wv * 16 + n15;
    int s = src[e], d = dst[e];
    float v = x[(size_t)s * 128 + n] + acc[r] + be[n];
    v = fmaxf(v, 0.f) + 1e-7f;
    v = fast_pow_pos(fminf(fmaxf(v, 1e-7f), 100.f), p);
    atomicAdd(&agg[(size_t)d * 128 + n], v);
  }
  if (wv == 0 && n15 == 0) {
#pragma unroll
    for (int r = 0; r < 8; ++r) atomicAdd(&cnt[dst[ebase + r + hf * 8]], 1.0f);
  }
}

// ------- node finalize: powermean root, MessageNorm, residual -> f16 -------
__global__ void node_fin_kernel(const float* __restrict__ agg, const float* __restrict__ cnt,
                                const float* __restrict__ x, const float* __restrict__ p_pow,
                                const float* __restrict__ scale, _Float16* __restrict__ outh) {
  int i = blockIdx.x, t = threadIdx.x;
  float c = fmaxf(cnt[i], 1.0f);
  float a = agg[(size_t)i * 128 + t] / c;
  a = fast_pow_pos(fminf(fmaxf(a, 1e-7f), 100.f), 1.0f / p_pow[0]);
  __shared__ float red[128];
  red[t] = a * a; __syncthreads();
  for (int s = 64; s > 0; s >>= 1) { if (t < s) red[t] += red[t + s]; __syncthreads(); }
  float na = fmaxf(sqrtf(red[0]), 1e-12f);
  __syncthreads();
  float xv = x[(size_t)i * 128 + t];
  red[t] = xv * xv; __syncthreads();
  for (int s = 64; s > 0; s >>= 1) { if (t < s) red[t] += red[t + s]; __syncthreads(); }
  float nx = sqrtf(red[0]);
  float o = (a / na) * nx * scale[0] + xv;
  outh[(size_t)i * 128 + t] = (_Float16)o;
}

// ------- GEMM + bias + LayerNorm + relu (+ optional w3-dot/sigmoid) --------
template <int W, int MODE>
__global__ void gemm_ln_kernel(const _Float16* __restrict__ A, const _Float16* __restrict__ BT,
                               const float* __restrict__ bias, const float* __restrict__ g,
                               const float* __restrict__ beta, _Float16* __restrict__ outH,
                               const float* __restrict__ w3, const float* __restrict__ wb3,
                               float* __restrict__ wout, int K) {
  constexpr int NW = W / 16;
  __shared__ float tile[16 * W];
  int lane = threadIdx.x & 31, hf = lane >> 4, n15 = lane & 15;
  int wv = threadIdx.x >> 5;
  int base = blockIdx.x * 16;
  const _Float16* arow = A + (size_t)(base + n15) * K;
  const _Float16* brow = BT + (size_t)(wv * 16 + n15) * K;
  v8f acc = {0.f, 0.f, 0.f, 0.f, 0.f, 0.f, 0.f, 0.f};
  for (int k = 0; k < K; k += 32)
    acc = wmma16(frag_a(arow + k, hf), frag_b(brow + k, hf), acc);
  int ncol = wv * 16 + n15;
#pragma unroll
  for (int r = 0; r < 8; ++r) tile[(r + hf * 8) * W + ncol] = acc[r] + bias[ncol];
  __syncthreads();
  constexpr int RPW = 16 / NW;
#pragma unroll
  for (int rr = 0; rr < RPW; ++rr) {
    int row = wv * RPW + rr;
    float s = 0.f, s2 = 0.f;
    for (int c = lane; c < W; c += 32) { float v = tile[row * W + c]; s += v; s2 += v * v; }
    for (int m = 16; m > 0; m >>= 1) { s += __shfl_xor(s, m, 32); s2 += __shfl_xor(s2, m, 32); }
    float mean = s / W, var = s2 / W - mean * mean;
    float inv = rsqrtf(var + 1e-5f);
    if (MODE == 0) {
      for (int c = lane; c < W; c += 32) {
        float v = fmaxf((tile[row * W + c] - mean) * inv * g[c] + beta[c], 0.f);
        outH[(size_t)(base + row) * W + c] = (_Float16)v;
      }
    } else {
      float d = 0.f;
      for (int c = lane; c < W; c += 32) {
        float v = fmaxf((tile[row * W + c] - mean) * inv * g[c] + beta[c], 0.f);
        d += v * w3[c];
      }
      for (int m = 16; m > 0; m >>= 1) d += __shfl_xor(d, m, 32);
      if (lane == 0) wout[base + row] = 1.f / (1.f + __expf(-(d + wb3[0])));
    }
  }
}

// ------------------- generic GEMM + bias (+relu), f32/f16 out --------------
__global__ void gemm_kernel(const _Float16* __restrict__ A, const _Float16* __restrict__ BT,
                            const float* __restrict__ bias, float* __restrict__ outF,
                            _Float16* __restrict__ outH, int M, int Nc, int K, int relu) {
  int lane = threadIdx.x & 31, hf = lane >> 4, n15 = lane & 15;
  int wid = (blockIdx.x * blockDim.x + threadIdx.x) >> 5;
  int nwaves = (gridDim.x * blockDim.x) >> 5;
  int tilesN = Nc >> 4;
  int tiles = (M >> 4) * tilesN;
  for (int t = wid; t < tiles; t += nwaves) {
    int tm = t / tilesN, tn = t - tm * tilesN;
    const _Float16* arow = A + (size_t)(tm * 16 + n15) * K;
    const _Float16* brow = BT + (size_t)(tn * 16 + n15) * K;
    v8f acc = {0.f, 0.f, 0.f, 0.f, 0.f, 0.f, 0.f, 0.f};
    for (int k = 0; k < K; k += 32)
      acc = wmma16(frag_a(arow + k, hf), frag_b(brow + k, hf), acc);
#pragma unroll
    for (int r = 0; r < 8; ++r) {
      int m = tm * 16 + r + hf * 8;
      int n = tn * 16 + n15;
      float v = acc[r] + bias[n];
      if (relu) v = fmaxf(v, 0.f);
      size_t o = (size_t)m * Nc + n;
      if (outF) outF[o] = v;
      if (outH) outH[o] = (_Float16)v;
    }
  }
}

// ------------------- weight_net stage 1: tiny K=2 + LN + relu --------------
__global__ void wn1_kernel(const float* __restrict__ aux, const float* __restrict__ w1,
                           const float* __restrict__ wb1, const float* __restrict__ wg1,
                           const float* __restrict__ wbt1, _Float16* __restrict__ h1h) {
  int g = threadIdx.x >> 7, c = threadIdx.x & 127;
  int e = blockIdx.x * 2 + g;
  float a0 = aux[(size_t)e * 2], a1 = aux[(size_t)e * 2 + 1];
  float v = a0 * w1[c] + a1 * w1[128 + c] + wb1[c];
  __shared__ float red[2][128];
  red[g][c] = v; __syncthreads();
  for (int s = 64; s > 0; s >>= 1) { if (c < s) red[g][c] += red[g][c + s]; __syncthreads(); }
  float mean = red[g][0] * (1.f / 128.f);
  __syncthreads();
  float dv = v - mean;
  red[g][c] = dv * dv; __syncthreads();
  for (int s = 64; s > 0; s >>= 1) { if (c < s) red[g][c] += red[g][c + s]; __syncthreads(); }
  float var = red[g][0] * (1.f / 128.f);
  float o = fmaxf(dv * rsqrtf(var + 1e-5f) * wg1[c] + wbt1[c], 0.f);
  h1h[(size_t)e * 128 + c] = (_Float16)o;
}

// ------------------- fusion: segment_max via monotonic u32 -----------------
__global__ void fusion_init_kernel(unsigned* __restrict__ enc) {
  int i = blockIdx.x * blockDim.x + threadIdx.x;
  if (i < 64 * 256) enc[i] = 0x007FFFFFu; // encoded -inf
}

// LDS-privatized segment max: each block reduces its edge slice into a
// 64-graph x 256-col LDS tile (ds_max_u32), then merges once into global.
__global__ void fusion_scatter_kernel(const float* __restrict__ x2, const int* __restrict__ src,
                                      const int* __restrict__ dst, const int* __restrict__ batch,
                                      const float* __restrict__ wedge, unsigned* __restrict__ enc) {
  __shared__ unsigned lenc[64 * 256];
  int t = threadIdx.x;
  for (int i = t; i < 64 * 256; i += 256) lenc[i] = 0x007FFFFFu;
  __syncthreads();
  for (int e = blockIdx.x; e < NE; e += gridDim.x) {
    int s = src[e], d = dst[e];
    float we = wedge[e];
    int b = batch[s];
    float v = (t < 128 ? x2[(size_t)s * 128 + t] : x2[(size_t)d * 128 + (t - 128)]) * we;
    unsigned u = __float_as_uint(v);
    u = (u & 0x80000000u) ? ~u : (u | 0x80000000u);
    atomicMax(&lenc[(b << 8) + t], u);
  }
  __syncthreads();
  for (int i = t; i < 64 * 256; i += 256) {
    unsigned u = lenc[i];
    if (u != 0x007FFFFFu) atomicMax(&enc[i], u);
  }
}

__global__ void fusion_final_kernel(const unsigned* __restrict__ enc, const float* __restrict__ gn,
                                    const float* __restrict__ bnp, const float* __restrict__ Wf,
                                    const float* __restrict__ bf, float* __restrict__ out) {
  int b = blockIdx.x, t = threadIdx.x;
  unsigned u = enc[b * 256 + t];
  u = (u & 0x80000000u) ? (u & 0x7FFFFFFFu) : ~u;
  float v = __uint_as_float(u);
  __shared__ float row[256];
  __shared__ float red[256];
  red[t] = v; __syncthreads();
  for (int s = 128; s > 0; s >>= 1) { if (t < s) red[t] += red[t + s]; __syncthreads(); }
  float mean = red[0] * (1.f / 256.f);
  __syncthreads();
  float dv = v - mean;
  red[t] = dv * dv; __syncthreads();
  for (int s = 128; s > 0; s >>= 1) { if (t < s) red[t] += red[t + s]; __syncthreads(); }
  float var = red[0] * (1.f / 256.f);
  row[t] = dv * rsqrtf(var + 1e-5f) * gn[t] + bnp[t];
  __syncthreads();
  if (t < 64) {
    float s = bf[t];
    for (int k = 0; k < 256; ++k) s += row[k] * Wf[k * 64 + t];
    out[b * 64 + t] = 1.f / (1.f + __expf(-s));
  }
}

// ------------------- BatchNorm (over node dim) stats + apply ---------------
__global__ void col_stats_kernel(const float* __restrict__ y, float* __restrict__ sum,
                                 float* __restrict__ sumsq, int rows) {
  int gid = blockIdx.x * blockDim.x + threadIdx.x;
  int c = gid & 127;
  int rstride = (gridDim.x * blockDim.x) >> 7;
  float s = 0.f, s2 = 0.f;
  for (int r = gid >> 7; r < rows; r += rstride) {
    float v = y[(size_t)r * 128 + c];
    s += v; s2 += v * v;
  }
  atomicAdd(&sum[c], s);
  atomicAdd(&sumsq[c], s2);
}

__global__ void bn_apply_kernel(const float* __restrict__ y, const float* __restrict__ sum,
                                const float* __restrict__ sumsq, const float* __restrict__ g,
                                const float* __restrict__ b, _Float16* __restrict__ outH,
                                int rows) {
  int i = blockIdx.x * blockDim.x + threadIdx.x;
  if (i >= rows * 128) return;
  int c = i & 127;
  float inv_n = 1.f / rows;
  float m = sum[c] * inv_n;
  float var = sumsq[c] * inv_n - m * m;
  float v = (y[i] - m) * rsqrtf(var + 1e-5f) * g[c] + b[c];
  outH[i] = (_Float16)fmaxf(v, 0.f);
}

// ===========================================================================
extern "C" void kernel_launch(void* const* d_in, const int* in_sizes, int n_in,
                              void* d_out, int out_size, void* d_ws, size_t ws_size,
                              hipStream_t stream) {
  const float* x   = (const float*)d_in[0];
  const float* ta  = (const float*)d_in[1];
  const float* aux = (const float*)d_in[2];
  int pi = 3;
  auto P = [&]() { return (const float*)d_in[pi++]; };
  const float *Wt = P(), *bt = P(), *Wa = P(), *ba = P(), *ge = P(), *be = P();
  struct Conv { const float *We, *be, *p, *scale, *W1, *b1, *g1, *bt1, *W2, *b2; } cv[2];
  for (int c = 0; c < 2; ++c) {
    cv[c].We = P(); cv[c].be = P(); cv[c].p = P(); cv[c].scale = P();
    cv[c].W1 = P(); cv[c].b1 = P(); cv[c].g1 = P(); cv[c].bt1 = P();
    cv[c].W2 = P(); cv[c].b2 = P();
  }
  const float *w1 = P(), *wb1 = P(), *wg1 = P(), *wbt1 = P();
  const float *w2 = P(), *wb2 = P(), *wg2 = P(), *wbt2 = P();
  const float *w3 = P(), *wb3 = P();
  const float *gn = P(), *bnp = P();
  const float *Wf = P(), *bf = P();
  const float *Wc1 = P(), *bc1 = P(), *gc = P(), *bc = P();
  const float *Wc2 = P(), *bc2 = P();
  const float *Wr1 = P(), *br1 = P();
  const float *Wr2 = P(), *br2 = P();
  const float *Wr3 = P(), *br3 = P();
  const float *Wr4 = P(), *br4 = P();
  const int* eidx  = (const int*)d_in[pi++];
  const int* src   = eidx;
  const int* dst   = eidx + NE;
  const int* batch = (const int*)d_in[pi++];

  // output slices: x2 | fusion | contr | recon
  float* out_x2     = (float*)d_out;
  float* out_fusion = out_x2 + (size_t)NN * 128;
  float* out_contr  = out_fusion + 64 * 64;
  float* out_recon  = out_contr + (size_t)NN * 128;

  // workspace carve
  char* wsb = (char*)d_ws;
  size_t off = 0;
  auto carve = [&](size_t bytes) -> void* {
    void* p = wsb + off;
    off = (off + bytes + 255) & ~(size_t)255;
    return p;
  };
  _Float16* efh  = (_Float16*)carve((size_t)NE * 64 * 2);
  _Float16* h1h  = (_Float16*)carve((size_t)NE * 128 * 2);
  float*    agg  = (float*)carve((size_t)NN * 128 * 4);
  float*    cnt  = (float*)carve((size_t)NN * 4);
  _Float16* outh = (_Float16*)carve((size_t)NN * 128 * 2);
  _Float16* hh   = (_Float16*)carve((size_t)NN * 256 * 2);
  float*    x1   = (float*)carve((size_t)NN * 128 * 4);
  _Float16* x2h  = (_Float16*)carve((size_t)NN * 128 * 2);
  float*    wedge = (float*)carve((size_t)NE * 4);
  float*    yb   = (float*)carve((size_t)NN * 128 * 4);
  _Float16* ch   = (_Float16*)carve((size_t)NN * 128 * 2);
  _Float16* t1h  = (_Float16*)carve((size_t)NN * 128 * 2);
  _Float16* ench = (_Float16*)carve((size_t)NN * 64 * 2);
  _Float16* t2h  = (_Float16*)carve((size_t)NN * 128 * 2);
  unsigned* fenc = (unsigned*)carve(64 * 256 * 4);
  float*    csum = (float*)carve(128 * 4);
  float*    csq  = (float*)carve(128 * 4);
  _Float16* WeT[2], *W1T[2], *W2T[2];
  for (int c = 0; c < 2; ++c) {
    WeT[c] = (_Float16*)carve(64 * 128 * 2);
    W1T[c] = (_Float16*)carve(128 * 256 * 2);
    W2T[c] = (_Float16*)carve(256 * 128 * 2);
  }
  _Float16* w2T  = (_Float16*)carve(128 * 128 * 2);
  _Float16* Wc1T = (_Float16*)carve(128 * 128 * 2);
  _Float16* Wc2T = (_Float16*)carve(128 * 128 * 2);
  _Float16* Wr1T = (_Float16*)carve(128 * 128 * 2);
  _Float16* Wr2T = (_Float16*)carve(128 * 64 * 2);
  _Float16* Wr3T = (_Float16*)carve(64 * 128 * 2);
  _Float16* Wr4T = (_Float16*)carve(128 * 128 * 2);

  auto wconv = [&](const float* W, _Float16* WT, int K, int Nc) {
    int n = K * Nc;
    wconv_kernel<<<(n + 255) / 256, 256, 0, stream>>>(W, WT, K, Nc);
  };
  for (int c = 0; c < 2; ++c) {
    wconv(cv[c].We, WeT[c], 64, 128);
    wconv(cv[c].W1, W1T[c], 128, 256);
    wconv(cv[c].W2, W2T[c], 256, 128);
  }
  wconv(w2, w2T, 128, 128);
  wconv(Wc1, Wc1T, 128, 128);
  wconv(Wc2, Wc2T, 128, 128);
  wconv(Wr1, Wr1T, 128, 128);
  wconv(Wr2, Wr2T, 128, 64);
  wconv(Wr3, Wr3T, 64, 128);
  wconv(Wr4, Wr4T, 128, 128);

  auto gemm = [&](const _Float16* A, const _Float16* BT, const float* bias,
                  float* oF, _Float16* oH, int M, int Nc, int K, int relu) {
    int tiles = (M >> 4) * (Nc >> 4);
    int blocks = (tiles + 7) / 8;
    gemm_kernel<<<blocks, 256, 0, stream>>>(A, BT, bias, oF, oH, M, Nc, K, relu);
  };

  // edge features
  ef_kernel<<<NE / 4, 256, 0, stream>>>(ta, aux, Wt, bt, Wa, ba, ge, be, efh);

  // two GENConv layers
  const float* xin = x;
  for (int c = 0; c < 2; ++c) {
    hipMemsetAsync(agg, 0, (size_t)NN * 128 * 4, stream);
    hipMemsetAsync(cnt, 0, (size_t)NN * 4, stream);
    msg_agg_kernel<<<NE / 16, 256, 0, stream>>>(efh, WeT[c], cv[c].be, xin, src, dst,
                                                cv[c].p, agg, cnt);
    node_fin_kernel<<<NN, 128, 0, stream>>>(agg, cnt, xin, cv[c].p, cv[c].scale, outh);
    gemm_ln_kernel<256, 0><<<NN / 16, 512, 0, stream>>>(outh, W1T[c], cv[c].b1, cv[c].g1,
                                                        cv[c].bt1, hh, nullptr, nullptr,
                                                        nullptr, 128);
    if (c == 0) {
      gemm(hh, W2T[c], cv[c].b2, x1, nullptr, NN, 128, 256, 0);
      xin = x1;
    } else {
      gemm(hh, W2T[c], cv[c].b2, out_x2, x2h, NN, 128, 256, 0);
    }
  }

  // weight_net
  wn1_kernel<<<NE / 2, 256, 0, stream>>>(aux, w1, wb1, wg1, wbt1, h1h);
  gemm_ln_kernel<128, 1><<<NE / 16, 256, 0, stream>>>(h1h, w2T, wb2, wg2, wbt2, nullptr,
                                                      w3, wb3, wedge, 128);

  // fusion: LDS-privatized segment max + LN + Wf + sigmoid
  fusion_init_kernel<<<64, 256, 0, stream>>>(fenc);
  fusion_scatter_kernel<<<512, 256, 0, stream>>>(out_x2, src, dst, batch, wedge, fenc);
  fusion_final_kernel<<<64, 256, 0, stream>>>(fenc, gn, bnp, Wf, bf, out_fusion);

  // contrastive head: BN over node dim
  gemm(x2h, Wc1T, bc1, yb, nullptr, NN, 128, 128, 0);
  hipMemsetAsync(csum, 0, 128 * 4, stream);
  hipMemsetAsync(csq, 0, 128 * 4, stream);
  col_stats_kernel<<<256, 256, 0, stream>>>(yb, csum, csq, NN);
  bn_apply_kernel<<<(NN * 128 + 255) / 256, 256, 0, stream>>>(yb, csum, csq, gc, bc, ch, NN);
  gemm(ch, Wc2T, bc2, out_contr, nullptr, NN, 128, 128, 0);

  // reconstruction head
  gemm(x2h, Wr1T, br1, nullptr, t1h, NN, 128, 128, 1);
  gemm(t1h, Wr2T, br2, nullptr, ench, NN, 64, 128, 0);
  gemm(ench, Wr3T, br3, nullptr, t2h, NN, 128, 64, 1);
  gemm(t2h, Wr4T, br4, out_recon, nullptr, NN, 128, 128, 0);
}
